// MixtureOfExperts_43696997269562
// MI455X (gfx1250) — matrix operand order repaired and checked by reading, not compiled, not run on hardware
//
#include <hip/hip_runtime.h>
#include <hip/hip_bf16.h>

// ---------------- problem constants ----------------
constexpr int D_   = 512;
constexpr int H_   = 2048;
constexpr int E_   = 8;
constexpr int T_   = 4096;   // 2 * 2048 tokens
constexpr int CAP  = 4096;   // worst-case tokens per expert
constexpr int TK_  = T_ * 2; // total routed rows (K=2)

typedef __attribute__((ext_vector_type(16))) __bf16 v16bf;
typedef __attribute__((ext_vector_type(8)))  __bf16 v8bf;
typedef __attribute__((ext_vector_type(8)))  float  v8f;

// ---------------- tiny init ----------------
__global__ void moe_init_kernel(int* counts) {
    if (threadIdx.x < E_) counts[threadIdx.x] = 0;
}

// ---------------- weight convert fp32 [e][k][n] -> bf16 [e][n][k] ----------------
__global__ __launch_bounds__(256)
void moe_convert_w_kernel(const float* __restrict__ W, __bf16* __restrict__ Wt,
                          int Kdim, int N) {
    size_t per   = (size_t)Kdim * N;
    size_t total = per * E_;
    size_t idx   = (size_t)blockIdx.x * 256 + threadIdx.x;
    if (idx >= total) return;
    size_t e = idx / per;
    size_t r = idx % per;
    size_t n = r / Kdim;
    size_t k = r % Kdim;
    Wt[idx] = (__bf16)W[e * per + k * (size_t)N + n];   // coalesced write, strided read
}

// ---------------- router: gate MLP + softmax + top-2 ----------------
__device__ __forceinline__ float gelu_exact(float v) {
    return 0.5f * v * (1.0f + erff(v * 0.7071067811865475f));
}

__global__ __launch_bounds__(128)
void moe_router_kernel(const float* __restrict__ x,
                       const float* __restrict__ Wg1, const float* __restrict__ bg1,
                       const float* __restrict__ Wg2, const float* __restrict__ bg2,
                       int* __restrict__ counts, int* __restrict__ expert_tok,
                       int* __restrict__ tok_e, int* __restrict__ tok_slot,
                       float* __restrict__ tok_g) {
    int wave = threadIdx.x >> 5;
    int lane = threadIdx.x & 31;
    int t = blockIdx.x * 4 + wave;
    if (t >= T_) return;
    const float* xt = x + (size_t)t * D_;

    float acc[16];
    #pragma unroll
    for (int j = 0; j < 16; ++j) acc[j] = 0.0f;
    for (int d = lane; d < D_; d += 32) {
        float xv = xt[d];
        const float* w = Wg1 + (size_t)d * 16;
        #pragma unroll
        for (int j = 0; j < 16; ++j) acc[j] += xv * w[j];
    }
    #pragma unroll
    for (int off = 16; off > 0; off >>= 1)
        #pragma unroll
        for (int j = 0; j < 16; ++j) acc[j] += __shfl_xor(acc[j], off, 32);

    if (lane == 0) {
        float h[16];
        #pragma unroll
        for (int j = 0; j < 16; ++j) h[j] = gelu_exact(acc[j] + bg1[j]);
        float lg[E_];
        #pragma unroll
        for (int e = 0; e < E_; ++e) {
            float s = bg2[e];
            #pragma unroll
            for (int j = 0; j < 16; ++j) s += h[j] * Wg2[j * E_ + e];
            lg[e] = s;
        }
        float mx = lg[0];
        #pragma unroll
        for (int e = 1; e < E_; ++e) mx = fmaxf(mx, lg[e]);
        float p[E_], sum = 0.0f;
        #pragma unroll
        for (int e = 0; e < E_; ++e) { p[e] = expf(lg[e] - mx); sum += p[e]; }
        float inv = 1.0f / sum;
        #pragma unroll
        for (int e = 0; e < E_; ++e) p[e] *= inv;
        // top-2 (ties -> lowest index, matches top_k)
        int i1 = 0;
        #pragma unroll
        for (int e = 1; e < E_; ++e) if (p[e] > p[i1]) i1 = e;
        int i2 = (i1 == 0) ? 1 : 0;
        #pragma unroll
        for (int e = 0; e < E_; ++e) if (e != i1 && p[e] > p[i2]) i2 = e;
        float rn = 1.0f / (p[i1] + p[i2]);

        int s1 = atomicAdd(&counts[i1], 1);
        int s2 = atomicAdd(&counts[i2], 1);
        expert_tok[i1 * CAP + s1] = t;
        expert_tok[i2 * CAP + s2] = t;
        tok_e[t * 2 + 0] = i1;  tok_slot[t * 2 + 0] = s1;  tok_g[t * 2 + 0] = p[i1] * rn;
        tok_e[t * 2 + 1] = i2;  tok_slot[t * 2 + 1] = s2;  tok_g[t * 2 + 1] = p[i2] * rn;
    }
}

// ---------------- 8-wide exclusive scan ----------------
__global__ void moe_scan_kernel(const int* __restrict__ counts, int* __restrict__ offsets) {
    if (threadIdx.x == 0) {
        int s = 0;
        for (int e = 0; e < E_; ++e) { offsets[e] = s; s += counts[e]; }
    }
}

// ---------------- gather routed tokens -> bf16 A ----------------
__global__ __launch_bounds__(256)
void moe_gather_kernel(const float* __restrict__ x, __bf16* __restrict__ Ag,
                       const int* __restrict__ counts, const int* __restrict__ offsets,
                       const int* __restrict__ expert_tok) {
    int e    = blockIdx.y;
    int slot = blockIdx.x * 4 + (threadIdx.x >> 6);
    int lane = threadIdx.x & 63;
    if (slot >= counts[e]) return;
    int row = offsets[e] + slot;
    int tok = expert_tok[e * CAP + slot];
    const float* src = x + (size_t)tok * D_;
    __bf16* dst = Ag + (size_t)row * D_;
    for (int i = lane; i < D_; i += 64) dst[i] = (__bf16)src[i];
}

// ---------------- grouped bf16 WMMA GEMM: C = A @ Wt^T + bias ----------------
// 64x64x64 block tile, 4 waves; per LDS stage each wave issues 8 WMMAs
// (2 K-substeps x 4 M-subtiles, one shared B fragment per substep).
// Fragments for a substep are batch-loaded BEFORE the 4 WMMAs so DS returns
// overlap compute (partial s_wait_dscnt instead of 0 before every WMMA).
// Double-buffered LDS with register staging: ONE barrier per stage.
#define MT 64
#define NT 64
#define KT 64
#define LDK (KT + 8)   // 72 elems / 144B row: 16B aligned chunks, conflict-free lanes

__global__ __launch_bounds__(128)
void moe_gemm_kernel(const __bf16* __restrict__ A, const __bf16* __restrict__ Wt,
                     const float* __restrict__ bias, float* __restrict__ C,
                     const int* __restrict__ counts, const int* __restrict__ offsets,
                     int Kdim, int N) {
    int e   = blockIdx.z;
    int cnt = counts[e];
    int m0  = blockIdx.y * MT;
    if (m0 >= cnt) return;
    int row0 = offsets[e] + m0;
    int n0   = blockIdx.x * NT;

    __shared__ __bf16 Asl[2][MT][LDK];
    __shared__ __bf16 Bsl[2][NT][LDK];

    int tid  = threadIdx.x;
    int wave = tid >> 5;
    int lane = tid & 31;
    int half = lane >> 4;
    int lr   = lane & 15;

    v8f acc[4] = {};
    const __bf16* We = Wt + (size_t)e * N * Kdim;

    // per-thread staging coordinates: 4 x v8bf chunks for A and for B
    int crow[4], ckc[4];
    #pragma unroll
    for (int l = 0; l < 4; ++l) {
        int idx = tid + 128 * l;       // 0..511 chunk id over 64 rows x 8 chunks
        crow[l] = idx >> 3;
        ckc[l]  = (idx & 7) * 8;
    }

    int nstages = Kdim / KT;

    // prologue: stage 0 -> LDS buffer 0
    #pragma unroll
    for (int l = 0; l < 4; ++l) {
        v8bf va = {};
        if (m0 + crow[l] < cnt)
            va = *(const v8bf*)(A + (size_t)(row0 + crow[l]) * Kdim + ckc[l]);
        *(v8bf*)(&Asl[0][crow[l]][ckc[l]]) = va;
        *(v8bf*)(&Bsl[0][crow[l]][ckc[l]]) =
            *(const v8bf*)(We + (size_t)(n0 + crow[l]) * Kdim + ckc[l]);
    }
    __syncthreads();

    for (int st = 0; st < nstages; ++st) {
        int cur = st & 1;
        bool have_next = (st + 1 < nstages);
        int k0n = (st + 1) * KT;

        // issue global loads for the next stage early (register staging)
        v8bf na[4], nb[4];
        if (have_next) {
            #pragma unroll
            for (int l = 0; l < 4; ++l) {
                v8bf va = {};
                if (m0 + crow[l] < cnt)
                    va = *(const v8bf*)(A + (size_t)(row0 + crow[l]) * Kdim + k0n + ckc[l]);
                na[l] = va;
                nb[l] = *(const v8bf*)(We + (size_t)(n0 + crow[l]) * Kdim + k0n + ckc[l]);
            }
        }

        // compute on LDS[cur]: 2 K-substeps x 4 M-subtiles
        #pragma unroll
        for (int kk = 0; kk < KT; kk += 32) {
            // Batch-load ALL fragments for this substep first (10 ds_load_b128),
            // then issue the 4 WMMAs; in-order DS returns let later loads hide
            // under earlier matrix ops.
            v16bf bfrag;
            v16bf afrag[4];
            {
                // B frag: N = lane&15; lane<16 -> K kk+0..15, lane>=16 -> K kk+16..31
                int bn = wave * 16 + lr;
                int kb = kk + half * 16;
                v8bf lo = *(const v8bf*)(&Bsl[cur][bn][kb]);
                v8bf hi = *(const v8bf*)(&Bsl[cur][bn][kb + 8]);
                #pragma unroll
                for (int i = 0; i < 8; ++i) { bfrag[i] = lo[i]; bfrag[i + 8] = hi[i]; }
            }
            #pragma unroll
            for (int s = 0; s < 4; ++s) {
                // A frag: lane<16 -> K kk+0..15; lane>=16 -> K kk+16..31
                int ar = s * 16 + lr;
                int ka = kk + half * 16;
                v8bf lo = *(const v8bf*)(&Asl[cur][ar][ka]);
                v8bf hi = *(const v8bf*)(&Asl[cur][ar][ka + 8]);
                #pragma unroll
                for (int i = 0; i < 8; ++i) { afrag[s][i] = lo[i]; afrag[s][i + 8] = hi[i]; }
            }
            #pragma unroll
            for (int s = 0; s < 4; ++s) {
                acc[s] = __builtin_amdgcn_wmma_f32_16x16x32_bf16(
                    false, afrag[s], false, bfrag, (short)0, acc[s], false, false);
            }
        }

        // store next stage into the other buffer (no conflict with current reads),
        // then one barrier makes it visible for the next iteration.
        if (have_next) {
            int nxt = cur ^ 1;
            #pragma unroll
            for (int l = 0; l < 4; ++l) {
                *(v8bf*)(&Asl[nxt][crow[l]][ckc[l]]) = na[l];
                *(v8bf*)(&Bsl[nxt][crow[l]][ckc[l]]) = nb[l];
            }
            __syncthreads();
        }
    }

    // epilogue: C/D layout: lane<16 -> N=lane,  M = v ; lane>=16 -> N=lane-16, M = 8+v
    int nc = n0 + wave * 16 + lr;
    float bv = bias[(size_t)e * N + nc];
    #pragma unroll
    for (int s = 0; s < 4; ++s) {
        #pragma unroll
        for (int v = 0; v < 8; ++v) {
            int m = s * 16 + half * 8 + v;
            if (m0 + m < cnt)
                C[(size_t)(row0 + m) * N + nc] = acc[s][v] + bv;
        }
    }
}

// ---------------- LayerNorm + GELU over full H row ----------------
__global__ __launch_bounds__(256)
void moe_ln_gelu_kernel(const float* __restrict__ Hpre,
                        const float* __restrict__ gamma, const float* __restrict__ beta,
                        __bf16* __restrict__ Hact,
                        const int* __restrict__ counts, const int* __restrict__ offsets) {
    int e = blockIdx.y;
    int slot = blockIdx.x;
    if (slot >= counts[e]) return;
    int row = offsets[e] + slot;
    const float* src = Hpre + (size_t)row * H_;

    float s = 0.0f, s2 = 0.0f;
    for (int i = threadIdx.x; i < H_; i += 256) {
        float v = src[i]; s += v; s2 += v * v;
    }
    __shared__ float red0[8], red1[8];
    #pragma unroll
    for (int off = 16; off > 0; off >>= 1) {
        s  += __shfl_xor(s,  off, 32);
        s2 += __shfl_xor(s2, off, 32);
    }
    int wave = threadIdx.x >> 5, lane = threadIdx.x & 31;
    if (lane == 0) { red0[wave] = s; red1[wave] = s2; }
    __syncthreads();
    if (wave == 0) {
        s  = (lane < 8) ? red0[lane] : 0.0f;
        s2 = (lane < 8) ? red1[lane] : 0.0f;
        #pragma unroll
        for (int off = 4; off > 0; off >>= 1) {
            s  += __shfl_xor(s,  off, 32);
            s2 += __shfl_xor(s2, off, 32);
        }
        if (lane == 0) { red0[0] = s; red1[0] = s2; }
    }
    __syncthreads();
    float mean = red0[0] * (1.0f / H_);
    float var  = red1[0] * (1.0f / H_) - mean * mean;
    float inv  = rsqrtf(var + 1e-5f);

    const float* g = gamma + (size_t)e * H_;
    const float* b = beta  + (size_t)e * H_;
    __bf16* dst = Hact + (size_t)row * H_;
    for (int i = threadIdx.x; i < H_; i += 256) {
        float v = (src[i] - mean) * inv * g[i] + b[i];
        dst[i] = (__bf16)gelu_exact(v);
    }
}

// ---------------- final combine: out[t] = sum_k g_k * O[row_k] ----------------
__global__ __launch_bounds__(128)
void moe_combine_kernel(const float* __restrict__ O,
                        const int* __restrict__ tok_e, const int* __restrict__ tok_slot,
                        const float* __restrict__ tok_g, const int* __restrict__ offsets,
                        float* __restrict__ out) {
    int t = blockIdx.x;
    int e0 = tok_e[t * 2 + 0], e1 = tok_e[t * 2 + 1];
    int r0 = offsets[e0] + tok_slot[t * 2 + 0];
    int r1 = offsets[e1] + tok_slot[t * 2 + 1];
    float g0 = tok_g[t * 2 + 0], g1 = tok_g[t * 2 + 1];
    const float* o0 = O + (size_t)r0 * D_;
    const float* o1 = O + (size_t)r1 * D_;
    float* dst = out + (size_t)t * D_;
    for (int i = threadIdx.x; i < D_; i += 128)
        dst[i] = g0 * o0[i] + g1 * o1[i];
}

// ---------------- launch ----------------
extern "C" void kernel_launch(void* const* d_in, const int* in_sizes, int n_in,
                              void* d_out, int out_size, void* d_ws, size_t ws_size,
                              hipStream_t stream) {
    const float* x   = (const float*)d_in[0];
    const float* Wg1 = (const float*)d_in[1];
    const float* bg1 = (const float*)d_in[2];
    const float* Wg2 = (const float*)d_in[3];
    const float* bg2 = (const float*)d_in[4];
    const float* W1  = (const float*)d_in[5];
    const float* b1  = (const float*)d_in[6];
    const float* g1  = (const float*)d_in[7];
    const float* be1 = (const float*)d_in[8];
    const float* W2  = (const float*)d_in[9];
    const float* b2  = (const float*)d_in[10];
    const float* g2  = (const float*)d_in[11];
    const float* be2 = (const float*)d_in[12];
    const float* W3  = (const float*)d_in[13];
    const float* b3  = (const float*)d_in[14];
    float* out = (float*)d_out;

    // workspace carve-out (256B aligned)
    char* p = (char*)d_ws;
    auto alloc = [&](size_t bytes) -> void* {
        void* r = (void*)p;
        p += (bytes + 255) & ~(size_t)255;
        return r;
    };
    int*    counts     = (int*)alloc(E_ * sizeof(int));
    int*    offsets    = (int*)alloc(E_ * sizeof(int));
    int*    expert_tok = (int*)alloc((size_t)E_ * CAP * sizeof(int));
    int*    tok_e      = (int*)alloc((size_t)T_ * 2 * sizeof(int));
    int*    tok_slot   = (int*)alloc((size_t)T_ * 2 * sizeof(int));
    float*  tok_g      = (float*)alloc((size_t)T_ * 2 * sizeof(float));
    __bf16* Ag         = (__bf16*)alloc((size_t)TK_ * D_ * sizeof(__bf16));
    float*  Hpre       = (float*)alloc((size_t)TK_ * H_ * sizeof(float));
    __bf16* Hact       = (__bf16*)alloc((size_t)TK_ * H_ * sizeof(__bf16));
    float*  O          = (float*)alloc((size_t)TK_ * D_ * sizeof(float));
    __bf16* Wt1        = (__bf16*)alloc((size_t)E_ * H_ * D_ * sizeof(__bf16));
    __bf16* Wt2        = (__bf16*)alloc((size_t)E_ * H_ * H_ * sizeof(__bf16));
    __bf16* Wt3        = (__bf16*)alloc((size_t)E_ * D_ * H_ * sizeof(__bf16));

    moe_init_kernel<<<1, 32, 0, stream>>>(counts);

    { // weight conversion (one-time per call; bf16 weight set ~100MB stays L2-resident)
        size_t t1 = (size_t)E_ * D_ * H_;
        moe_convert_w_kernel<<<(unsigned)((t1 + 255) / 256), 256, 0, stream>>>(W1, Wt1, D_, H_);
        size_t t2 = (size_t)E_ * H_ * H_;
        moe_convert_w_kernel<<<(unsigned)((t2 + 255) / 256), 256, 0, stream>>>(W2, Wt2, H_, H_);
        size_t t3 = (size_t)E_ * H_ * D_;
        moe_convert_w_kernel<<<(unsigned)((t3 + 255) / 256), 256, 0, stream>>>(W3, Wt3, H_, D_);
    }

    moe_router_kernel<<<T_ / 4, 128, 0, stream>>>(x, Wg1, bg1, Wg2, bg2,
                                                  counts, expert_tok, tok_e, tok_slot, tok_g);
    moe_scan_kernel<<<1, 32, 0, stream>>>(counts, offsets);
    moe_gather_kernel<<<dim3(CAP / 4, E_), 256, 0, stream>>>(x, Ag, counts, offsets, expert_tok);

    // FFN1: [rows x 512] @ [512 x 2048]
    moe_gemm_kernel<<<dim3(H_ / NT, CAP / MT, E_), 128, 0, stream>>>(
        Ag, Wt1, b1, Hpre, counts, offsets, D_, H_);
    moe_ln_gelu_kernel<<<dim3(CAP, E_), 256, 0, stream>>>(Hpre, g1, be1, Hact, counts, offsets);

    // FFN2: [rows x 2048] @ [2048 x 2048]
    moe_gemm_kernel<<<dim3(H_ / NT, CAP / MT, E_), 128, 0, stream>>>(
        Hact, Wt2, b2, Hpre, counts, offsets, H_, H_);
    moe_ln_gelu_kernel<<<dim3(CAP, E_), 256, 0, stream>>>(Hpre, g2, be2, Hact, counts, offsets);

    // FFN3: [rows x 2048] @ [2048 x 512]
    moe_gemm_kernel<<<dim3(D_ / NT, CAP / MT, E_), 128, 0, stream>>>(
        Hact, Wt3, b3, O, counts, offsets, H_, D_);

    moe_combine_kernel<<<T_, 128, 0, stream>>>(O, tok_e, tok_slot, tok_g, offsets, out);
}
